// FilmAttention_73289321939598
// MI455X (gfx1250) — matrix operand order repaired and checked by reading, not compiled, not run on hardware
//
#include <hip/hip_runtime.h>
#include <hip/hip_bf16.h>

typedef __attribute__((ext_vector_type(16))) _Float16 v16h;
typedef __attribute__((ext_vector_type(8)))  float    v8f;

// Problem constants
#define Bn 16
#define Fn 512
#define Tn 256
#define Cn 32
#define Zn 64
#define TCn 128
#define UHn 64   // U*H
#define Hn 4

// ---------------------------------------------------------------------------
// Kernel 1: cond resize + softmax_z + gamma/beta dots + km projection
// ---------------------------------------------------------------------------
__global__ __launch_bounds__(256) void k_prep(
    const float* __restrict__ cond,   // [B,Z,TC]
    const float* __restrict__ gam,    // [Z]
    const float* __restrict__ bet,    // [Z]
    const float* __restrict__ wk,     // [2,64]
    const float* __restrict__ wkb,    // [64]
    float* __restrict__ g_o,          // [B*T]
    float* __restrict__ bt_o,         // [B*T]
    float* __restrict__ km_o)         // [B*T,64]
{
    int r = blockIdx.x * blockDim.x + threadIdx.x;
    if (r >= Bn * Tn) return;
    int b = r >> 8, t = r & (Tn - 1);
    int st = t >> 1;  // nearest: floor((t+0.5)*TC/T) = t/2
    const float* cp = cond + (size_t)b * Zn * TCn + st;
    float m = -3.0e38f;
    for (int z = 0; z < Zn; ++z) m = fmaxf(m, cp[(size_t)z * TCn]);
    float s = 0.f, sg = 0.f, sb = 0.f;
    for (int z = 0; z < Zn; ++z) {
        float e = __expf(cp[(size_t)z * TCn] - m);
        s += e; sg += e * gam[z]; sb += e * bet[z];
    }
    float gv = sg / s, bv = sb / s;
    g_o[r] = gv; bt_o[r] = bv;
    float* km = km_o + (size_t)r * UHn;
    for (int n = 0; n < UHn; ++n)
        km[n] = gv * wk[n] + bv * wk[UHn + n] + wkb[n];
}

// ---------------------------------------------------------------------------
// Kernel 2a: one-time wq repack f32 -> f16 in WMMA B-fragment order.
// Layout: wqh[f][nt][lane][h], h=0..15; element = wq[(f*32 + K)*64 + n]
//   with K = (h&7) + 8*(lane>>4) + 16*(h>>3), n = nt*16 + (lane&15).
// Each lane's fragment is then a contiguous 32B global read (coalesced).
// ---------------------------------------------------------------------------
__global__ __launch_bounds__(256) void k_wq_pack(
    const float* __restrict__ wq,     // [16384,64]
    _Float16* __restrict__ wqh)       // [512*4*32*16]
{
    int idx = blockIdx.x * blockDim.x + threadIdx.x;   // 0 .. 1048575
    int h  = idx & 15;
    int L  = (idx >> 4) & 31;
    int nt = (idx >> 9) & 3;
    int f  = idx >> 11;
    int K  = (h & 7) + 8 * (L >> 4) + 16 * (h >> 3);
    int n  = nt * 16 + (L & 15);
    wqh[idx] = (_Float16)wq[(size_t)(f * 32 + K) * UHn + n];
}

// ---------------------------------------------------------------------------
// Kernel 2b: qm = x_perm[4096,16384] @ wq[16384,64] + bias (f16 WMMA, f32 acc)
// One wave per 16-row M tile, full N=64; 256 independent waves.
// A K-chunk of 32 == one contiguous x[b,f,t,0:32] row; B fragments are
// prepacked and contiguous -> no LDS, no barriers in the hot loop.
// ---------------------------------------------------------------------------
__global__ __launch_bounds__(32) void k_qm_gemm(
    const float* __restrict__ x,     // [B,F,T,C]
    const v16h*  __restrict__ wqh,   // packed B fragments [f][nt][lane]
    const float* __restrict__ wqb,   // [64]
    float* __restrict__ qm)          // [4096,64]
{
    const int lane = threadIdx.x & 31;
    const int l16  = lane & 15;
    const int lgrp = lane >> 4;           // 0: lanes 0-15, 1: lanes 16-31
    const int tile = blockIdx.x;          // 0..255 (M tiles of 16 rows)
    const int b    = tile >> 4;
    const int t0   = (tile & 15) * 16;
    const int row_t = t0 + l16;           // A layout: lane holds row M = lane%16

    // x base for (b, f=0, row_t, c=0); lane grp selects K halves
    const float* xrow = x + (((size_t)b * Fn) * Tn + row_t) * Cn;
    const int off0 = lgrp * 8;            // grp0: K {0..7,16..23}; grp1: {8..15,24..31}

    v8f acc[4] = {};

    for (int f = 0; f < Fn; ++f) {
        const float* xa = xrow + (size_t)f * Tn * Cn;
        // A fragment raw loads (16-bit A 16x32 layout)
        const float4 a0 = *(const float4*)(xa + off0);
        const float4 a1 = *(const float4*)(xa + off0 + 4);
        const float4 a2 = *(const float4*)(xa + off0 + 16);
        const float4 a3 = *(const float4*)(xa + off0 + 20);
        // B fragments: contiguous 32B per lane, consecutive across lanes
        const v16h b0 = wqh[(f * 4 + 0) * 32 + lane];
        const v16h b1 = wqh[(f * 4 + 1) * 32 + lane];
        const v16h b2 = wqh[(f * 4 + 2) * 32 + lane];
        const v16h b3 = wqh[(f * 4 + 3) * 32 + lane];
        // pull the A stream ahead while WMMAs run
        if (f + 8 < Fn) __builtin_prefetch(xa + (size_t)8 * Tn * Cn + off0, 0, 3);

        v16h af;
        af[0]=(_Float16)a0.x; af[1]=(_Float16)a0.y; af[2]=(_Float16)a0.z; af[3]=(_Float16)a0.w;
        af[4]=(_Float16)a1.x; af[5]=(_Float16)a1.y; af[6]=(_Float16)a1.z; af[7]=(_Float16)a1.w;
        af[8]=(_Float16)a2.x; af[9]=(_Float16)a2.y; af[10]=(_Float16)a2.z; af[11]=(_Float16)a2.w;
        af[12]=(_Float16)a3.x; af[13]=(_Float16)a3.y; af[14]=(_Float16)a3.z; af[15]=(_Float16)a3.w;

        acc[0] = __builtin_amdgcn_wmma_f32_16x16x32_f16(false, af, false, b0, (short)0, acc[0], false, false);
        acc[1] = __builtin_amdgcn_wmma_f32_16x16x32_f16(false, af, false, b1, (short)0, acc[1], false, false);
        acc[2] = __builtin_amdgcn_wmma_f32_16x16x32_f16(false, af, false, b2, (short)0, acc[2], false, false);
        acc[3] = __builtin_amdgcn_wmma_f32_16x16x32_f16(false, af, false, b3, (short)0, acc[3], false, false);
    }

    // store: C/D layout lane->N, vgpr v -> M = v + 8*lgrp
    const size_t rbase = (size_t)b * Tn + t0;
    #pragma unroll
    for (int nt = 0; nt < 4; ++nt) {
        const int n = nt * 16 + l16;
        const float bias = wqb[n];
        #pragma unroll
        for (int v = 0; v < 8; ++v) {
            const int M = v + 8 * lgrp;
            qm[(rbase + M) * UHn + n] = acc[nt][v] + bias;
        }
    }
}

// ---------------------------------------------------------------------------
// Kernel 3: attention per (b,h) with online softmax; Dv==1 so we only need
// running (sum_exp, sum_exp*g, sum_exp*bt). One workgroup per (b,h).
// ---------------------------------------------------------------------------
__global__ __launch_bounds__(256) void k_attn(
    const float* __restrict__ qm,   // [4096,64]
    const float* __restrict__ km,   // [4096,64]
    const float* __restrict__ g_i,  // [4096]
    const float* __restrict__ bt_i, // [4096]
    float* __restrict__ sg_o,       // [B,H,T]
    float* __restrict__ sb_o)       // [B,H,T]
{
    __shared__ float4 kh[Tn * 4];
    __shared__ float  gs[Tn], bs[Tn];
    const int b = blockIdx.x >> 2, h = blockIdx.x & 3;
    const int t = threadIdx.x;
    const size_t base = ((size_t)b * Tn + t) * UHn + h * 16;

    const float4* kmp = (const float4*)(km + base);
    kh[t * 4 + 0] = kmp[0]; kh[t * 4 + 1] = kmp[1];
    kh[t * 4 + 2] = kmp[2]; kh[t * 4 + 3] = kmp[3];
    gs[t] = g_i[b * Tn + t];
    bs[t] = bt_i[b * Tn + t];

    const float4* qmp = (const float4*)(qm + base);
    const float4 q0 = qmp[0], q1 = qmp[1], q2 = qmp[2], q3 = qmp[3];
    __syncthreads();

    float m = -3.0e38f, se = 0.f, sg = 0.f, sb = 0.f;
    for (int k = 0; k < Tn; ++k) {
        const float4 k0 = kh[k * 4 + 0], k1 = kh[k * 4 + 1];
        const float4 k2 = kh[k * 4 + 2], k3 = kh[k * 4 + 3];
        float s = q0.x*k0.x + q0.y*k0.y + q0.z*k0.z + q0.w*k0.w
                + q1.x*k1.x + q1.y*k1.y + q1.z*k1.z + q1.w*k1.w
                + q2.x*k2.x + q2.y*k2.y + q2.z*k2.z + q2.w*k2.w
                + q3.x*k3.x + q3.y*k3.y + q3.z*k3.z + q3.w*k3.w;
        s *= 0.25f;   // 1/sqrt(U)
        const float nm   = fmaxf(m, s);
        const float corr = __expf(m - nm);
        const float p    = __expf(s - nm);
        se = se * corr + p;
        sg = sg * corr + p * gs[k];
        sb = sb * corr + p * bs[k];
        m = nm;
    }
    const int o = (b * Hn + h) * Tn + t;
    sg_o[o] = sg / se;
    sb_o[o] = sb / se;
}

// ---------------------------------------------------------------------------
// Kernel 4: head-combine dense: g_out/b_out[b,t] = d_b + sum_h d_k[h]*s[b,h,t]
// ---------------------------------------------------------------------------
__global__ __launch_bounds__(256) void k_combine(
    const float* __restrict__ sg, const float* __restrict__ sb,
    const float* __restrict__ dk, const float* __restrict__ db,
    float* __restrict__ go, float* __restrict__ bo)
{
    int r = blockIdx.x * blockDim.x + threadIdx.x;
    if (r >= Bn * Tn) return;
    int b = r >> 8, t = r & (Tn - 1);
    float a = db[0], c = db[0];
    #pragma unroll
    for (int h = 0; h < Hn; ++h) {
        a += dk[h] * sg[(b * Hn + h) * Tn + t];
        c += dk[h] * sb[(b * Hn + h) * Tn + t];
    }
    go[r] = a; bo[r] = c;
}

// ---------------------------------------------------------------------------
// Kernel 5: FiLM epilogue out = b_out + x * g_out, float4 stream
// ---------------------------------------------------------------------------
__global__ __launch_bounds__(256) void k_film(
    const float* __restrict__ x,
    const float* __restrict__ go, const float* __restrict__ bo,
    float* __restrict__ out)
{
    const size_t j = (size_t)blockIdx.x * blockDim.x + threadIdx.x; // float4 idx
    const int t = (int)((j >> 3) & (Tn - 1));
    const int b = (int)(j >> 20);                 // j / (F*T*C/4)
    const float gv = go[b * Tn + t];
    const float bv = bo[b * Tn + t];
    const float4 xv = ((const float4*)x)[j];
    float4 o;
    o.x = bv + xv.x * gv; o.y = bv + xv.y * gv;
    o.z = bv + xv.z * gv; o.w = bv + xv.w * gv;
    ((float4*)out)[j] = o;
}

// ---------------------------------------------------------------------------
extern "C" void kernel_launch(void* const* d_in, const int* in_sizes, int n_in,
                              void* d_out, int out_size, void* d_ws, size_t ws_size,
                              hipStream_t stream) {
    const float* x    = (const float*)d_in[0];
    const float* cond = (const float*)d_in[1];
    const float* gam  = (const float*)d_in[2];
    const float* bet  = (const float*)d_in[3];
    const float* wq   = (const float*)d_in[4];
    const float* wqb  = (const float*)d_in[5];
    const float* wk   = (const float*)d_in[6];
    const float* wkb  = (const float*)d_in[7];
    const float* dk   = (const float*)d_in[8];
    const float* db   = (const float*)d_in[9];
    float* out = (float*)d_out;

    float* ws = (float*)d_ws;
    float* qm = ws;                        // 4096*64 f32
    float* km = ws + 262144;               // 4096*64 f32
    float* g  = ws + 524288;               // 4096
    float* bt = ws + 528384;               // 4096
    float* sg = ws + 532480;               // 16*4*256
    float* sb = ws + 548864;               // 16*4*256
    float* go = ws + 565248;               // 4096
    float* bo = ws + 569344;               // 4096
    _Float16* wqh = (_Float16*)(ws + 573440);  // 1048576 f16 = 2MB (32B aligned)

    k_prep<<<16, 256, 0, stream>>>(cond, gam, bet, wk, wkb, g, bt, km);
    k_wq_pack<<<4096, 256, 0, stream>>>(wq, wqh);
    k_qm_gemm<<<256, 32, 0, stream>>>(x, (const v16h*)wqh, wqb, qm);
    k_attn<<<Bn * Hn, 256, 0, stream>>>(qm, km, g, bt, sg, sb);
    k_combine<<<16, 256, 0, stream>>>(sg, sb, dk, db, go, bo);
    k_film<<<(Bn * Fn * Tn * Cn / 4) / 256, 256, 0, stream>>>(x, go, bo, out);
}
